// Dense_73890617361016
// MI455X (gfx1250) — matrix operand order repaired and checked by reading, not compiled, not run on hardware
//
#include <hip/hip_runtime.h>

// ---------------------------------------------------------------------------
// Dense layer (MPC fixed-point semantics): out = quantize(relu(x @ w + b))
//   x: [8192,1024] f32, w: [1024,1024] f32, b: [1024] f32
//   quantize(v) = rint(v * 2^16) / 2^16
//
// Strategy for MI455X (gfx1250, wave32, WMMA):
//   f16 two-limb split GEMM: x = xh + xl, w = wh + wl (f16 limbs),
//   acc += xh*wh + xh*wl + xl*wh  via  v_wmma_f32_16x16x32_f16 (fp32 accum).
//   ~22 mantissa bits per product + fp32 accumulation -> matches the fp32
//   reference at the 2^-16 output grid.
// ---------------------------------------------------------------------------

typedef __attribute__((ext_vector_type(16))) _Float16 v16h;
typedef __attribute__((ext_vector_type(8)))  _Float16 v8h;
typedef __attribute__((ext_vector_type(4)))  _Float16 v4h;
typedef __attribute__((ext_vector_type(8)))  float    v8f;

#define M_DIM 8192
#define N_DIM 1024
#define K_DIM 1024

#define BM 128
#define BN 128
#define BK 32
#define LDS_STRIDE (BK + 8)          // 40 halves = 80B row stride (16B aligned, conflict-free)

#define FXP_SCALE     65536.0f
#define FXP_INV_SCALE (1.0f / 65536.0f)

static __device__ __forceinline__ v16h cat16(v8h a, v8h b) {
  v16h r;
#pragma unroll
  for (int i = 0; i < 8; ++i) { r[i] = a[i]; r[i + 8] = b[i]; }
  return r;
}

__global__ __launch_bounds__(256) void dense_relu_fxp_kernel(
    const float* __restrict__ X,     // [M,K]
    const float* __restrict__ W,     // [K,N]
    const float* __restrict__ Bias,  // [N]
    float* __restrict__ Out)         // [M,N]
{
  __shared__ _Float16 sAh[BM][LDS_STRIDE];   // A tile, row-major [m][k], hi limb
  __shared__ _Float16 sAl[BM][LDS_STRIDE];   // lo limb
  __shared__ _Float16 sBh[BN][LDS_STRIDE];   // B tile, TRANSPOSED [n][k], hi limb
  __shared__ _Float16 sBl[BN][LDS_STRIDE];   // lo limb

  const int tid  = threadIdx.x;
  const int lane = tid & 31;
  const int wave = tid >> 5;
  const int wm   = wave & 3;   // 4 wave-rows  -> 32 M-rows each
  const int wn   = wave >> 2;  // 2 wave-cols  -> 64 N-cols each

  const int m0 = blockIdx.y * BM;
  const int n0 = blockIdx.x * BN;

  v8f acc[2][4];
#pragma unroll
  for (int tm = 0; tm < 2; ++tm)
#pragma unroll
    for (int tn = 0; tn < 4; ++tn) {
      v8f z = {0.f, 0.f, 0.f, 0.f, 0.f, 0.f, 0.f, 0.f};
      acc[tm][tn] = z;
    }

  // fragment addressing (per ISA 16-bit A/B VGPR layouts, wave32)
  const int frow  = lane & 15;
  const int koffA = (lane & 16) ? 8 : 0;    // A: lanes 16-31 hold K=8..15 / 24..31
  const int koffB = (lane & 16) ? 8 : 0;    // B: lanes 16-31 hold K=8..15 / 24..31

  for (int k0 = 0; k0 < K_DIM; k0 += BK) {
    // ---- stage A tile: 128x32 f32, 4 x float4 per thread ----
#pragma unroll
    for (int i = 0; i < 4; ++i) {
      int idx = tid + i * 256;              // 0..1023 float4 slots
      int row = idx >> 3;                   // 8 float4 per row
      int c   = (idx & 7) << 2;             // float column
      const float4 v = *(const float4*)(X + (size_t)(m0 + row) * K_DIM + k0 + c);
      float f[4] = {v.x, v.y, v.z, v.w};
      v4h hh, hl;
#pragma unroll
      for (int j = 0; j < 4; ++j) {
        _Float16 h = (_Float16)f[j];
        hh[j] = h;
        hl[j] = (_Float16)(f[j] - (float)h);
      }
      *(v4h*)&sAh[row][c] = hh;
      *(v4h*)&sAl[row][c] = hl;
    }

    // ---- stage B tile: 32x128 f32, stored transposed [n][k] ----
#pragma unroll
    for (int i = 0; i < 4; ++i) {
      int idx  = tid + i * 256;             // 0..1023 float4 slots
      int krow = idx >> 5;                  // 32 float4 per K-row
      int c    = (idx & 31) << 2;           // N column
      const float4 v = *(const float4*)(W + (size_t)(k0 + krow) * N_DIM + n0 + c);
      float f[4] = {v.x, v.y, v.z, v.w};
#pragma unroll
      for (int j = 0; j < 4; ++j) {
        _Float16 h = (_Float16)f[j];
        sBh[c + j][krow] = h;
        sBl[c + j][krow] = (_Float16)(f[j] - (float)h);
      }
    }

    // prefetch next K-slice while we compute (global_prefetch_b8)
    if (k0 + BK < K_DIM) {
      int idx = tid;
      int row = idx >> 3;
      int c   = (idx & 7) << 2;
      __builtin_prefetch(X + (size_t)(m0 + row) * K_DIM + (k0 + BK) + c, 0, 0);
      int krow = idx >> 5;
      int cn   = (idx & 31) << 2;
      __builtin_prefetch(W + (size_t)(k0 + BK + krow) * N_DIM + n0 + cn, 0, 0);
    }

    __syncthreads();

    // ---- A fragments: two 16B chunks per lane (K 0..7 & 16..23 / 8..15 & 24..31) ----
    v16h ah[2], al[2];
#pragma unroll
    for (int tm = 0; tm < 2; ++tm) {
      int row = wm * 32 + tm * 16 + frow;
      ah[tm] = cat16(*(const v8h*)&sAh[row][koffA],
                     *(const v8h*)&sAh[row][koffA + 16]);
      al[tm] = cat16(*(const v8h*)&sAl[row][koffA],
                     *(const v8h*)&sAl[row][koffA + 16]);
    }

    // ---- B fragments (two 16B chunks per lane in transposed layout) + WMMA ----
#pragma unroll
    for (int tn = 0; tn < 4; ++tn) {
      int col = wn * 64 + tn * 16 + frow;
      v16h bh = cat16(*(const v8h*)&sBh[col][koffB],
                      *(const v8h*)&sBh[col][koffB + 16]);
      v16h bl = cat16(*(const v8h*)&sBl[col][koffB],
                      *(const v8h*)&sBl[col][koffB + 16]);
#pragma unroll
      for (int tm = 0; tm < 2; ++tm) {
        acc[tm][tn] = __builtin_amdgcn_wmma_f32_16x16x32_f16(
            false, ah[tm], false, bh, (short)0, acc[tm][tn], false, false);
        acc[tm][tn] = __builtin_amdgcn_wmma_f32_16x16x32_f16(
            false, ah[tm], false, bl, (short)0, acc[tm][tn], false, false);
        acc[tm][tn] = __builtin_amdgcn_wmma_f32_16x16x32_f16(
            false, al[tm], false, bh, (short)0, acc[tm][tn], false, false);
      }
    }

    __syncthreads();
  }

  // ---- epilogue: bias + relu-sign + fixed-point re-quantization ----
  // C/D layout: VGPR r -> M = r (+8 for lanes 16-31), N = lane & 15
#pragma unroll
  for (int tn = 0; tn < 4; ++tn) {
    int col = n0 + wn * 64 + tn * 16 + frow;
    float bias = Bias[col];
#pragma unroll
    for (int tm = 0; tm < 2; ++tm) {
      int rbase = m0 + wm * 32 + tm * 16 + ((lane & 16) ? 8 : 0);
#pragma unroll
      for (int r = 0; r < 8; ++r) {
        float y = acc[tm][tn][r] + bias;
        float o = (y > 0.0f) ? rintf(y * FXP_SCALE) * FXP_INV_SCALE : 0.0f;
        Out[(size_t)(rbase + r) * N_DIM + col] = o;
      }
    }
  }
}

extern "C" void kernel_launch(void* const* d_in, const int* in_sizes, int n_in,
                              void* d_out, int out_size, void* d_ws, size_t ws_size,
                              hipStream_t stream) {
  (void)in_sizes; (void)n_in; (void)out_size; (void)d_ws; (void)ws_size;
  const float* X    = (const float*)d_in[0];
  const float* W    = (const float*)d_in[1];
  const float* Bias = (const float*)d_in[2];
  float* Out        = (float*)d_out;

  dim3 grid(N_DIM / BN, M_DIM / BM);   // 8 x 64 blocks
  dim3 block(256, 1, 1);               // 8 wave32s
  dense_relu_fxp_kernel<<<grid, block, 0, stream>>>(X, W, Bias, Out);
}
